// GIN_45681272160997
// MI455X (gfx1250) — matrix operand order repaired and checked by reading, not compiled, not run on hardware
//
#include <hip/hip_runtime.h>

// ---------------------------------------------------------------------------
// GIN layer for MI455X (gfx1250, wave32, WMMA bf16 16x16x32, f32 accumulate)
// ---------------------------------------------------------------------------

typedef __bf16 bf16_t;
typedef __attribute__((ext_vector_type(16))) __bf16 v16bf;
typedef __attribute__((ext_vector_type(8)))  float  v8f;

constexpr int    NNODES = 100000;
constexpr int    NEDGES = 600000;
constexpr int    HD     = 128;     // IN == H == 128
constexpr int    OD     = 64;      // OUT
constexpr int    WS     = 136;     // padded LDS stride (elements) for weights
constexpr int    TS     = 136;     // padded LDS stride (elements) for act tiles
constexpr float  BN_EPS = 1e-5f;

union Frag { v16bf v; uint4 q[2]; };

__device__ __forceinline__ unsigned pack2(float a, float b) {
  union { __bf16 h[2]; unsigned u; } p;
  p.h[0] = (__bf16)a; p.h[1] = (__bf16)b;
  return p.u;
}
__device__ __forceinline__ void unpack8(uint4 r, float* f) {
  union { unsigned u; __bf16 h[2]; } p;
  p.u = r.x; f[0] = (float)p.h[0]; f[1] = (float)p.h[1];
  p.u = r.y; f[2] = (float)p.h[0]; f[3] = (float)p.h[1];
  p.u = r.z; f[4] = (float)p.h[0]; f[5] = (float)p.h[1];
  p.u = r.w; f[6] = (float)p.h[0]; f[7] = (float)p.h[1];
}

// One 16x16 output tile (transposed: rows=out-channels, cols=node-rows),
// K=128 via 4 x wmma_f32_16x16x32_bf16.
// wrow = &Wt[(nt*16 + (lane&15)) * WS], hrow = &tile[(lane&15) * TS]
__device__ __forceinline__ v8f gemm_tile(const bf16_t* wrow, const bf16_t* hrow,
                                         int ka0, int kb0) {
  v8f c = {};
#pragma unroll
  for (int kt = 0; kt < 4; ++kt) {
    Frag a, b;
    const bf16_t* wp = wrow + kt * 32 + ka0;     // A: 16x32 bf16 fragment
    a.q[0] = *(const uint4*)wp;                  //   K = ka0 .. ka0+7
    a.q[1] = *(const uint4*)(wp + 16);           //   K = ka0+16 .. ka0+23
    const bf16_t* hp = hrow + kt * 32 + kb0;     // B: 32x16 bf16 fragment
    b.q[0] = *(const uint4*)hp;                  //   K = kb0 .. kb0+7
    b.q[1] = *(const uint4*)(hp + 16);           //   K = kb0+16 .. kb0+23
    c = __builtin_amdgcn_wmma_f32_16x16x32_bf16(false, a.v, false, b.v,
                                                (short)0, c, false, false);
  }
  return c;
}

// ---------------------------------------------------------------------------
// Kernel 0: agg = x  (float4 copy)
// ---------------------------------------------------------------------------
__global__ __launch_bounds__(256) void k_init(const float* __restrict__ x,
                                              float* __restrict__ agg) {
  size_t i = (size_t)blockIdx.x * 256 + threadIdx.x;   // N*32 float4 chunks
  ((float4*)agg)[i] = ((const float4*)x)[i];
}

__global__ __launch_bounds__(256) void k_zero(float* __restrict__ stats) {
  stats[threadIdx.x] = 0.0f;   // sum[128] + sumsq[128]
}

// ---------------------------------------------------------------------------
// Kernel 1: scatter-add  agg[dst] += x[src]  (thread = edge x float4 chunk)
// ---------------------------------------------------------------------------
__global__ __launch_bounds__(256) void k_scatter(const long long* __restrict__ ei,
                                                 const float* __restrict__ x,
                                                 float* __restrict__ agg) {
  size_t gid = (size_t)blockIdx.x * 256 + threadIdx.x;  // E*32 threads exactly
  int e  = (int)(gid >> 5);
  int c4 = (int)(gid & 31);
  long long s = ei[e];
  long long d = ei[NEDGES + e];
  float4 v = ((const float4*)x)[(size_t)s * 32 + c4];
  float* p = agg + (size_t)d * HD + c4 * 4;
  unsafeAtomicAdd(p + 0, v.x);
  unsafeAtomicAdd(p + 1, v.y);
  unsafeAtomicAdd(p + 2, v.z);
  unsafeAtomicAdd(p + 3, v.w);
}

// ---------------------------------------------------------------------------
// Kernel 2: h1 = agg @ W1 + b1 (bf16 out) + BN column sum / sumsq (f32)
// 256 threads = 8 waves, 16 rows per wave, 128 rows per block.
// ---------------------------------------------------------------------------
__global__ __launch_bounds__(256) void k_gemm1(const float* __restrict__ agg,
                                               const float* __restrict__ W1,
                                               const float* __restrict__ b1,
                                               bf16_t* __restrict__ h1,
                                               float* __restrict__ gsum,
                                               float* __restrict__ gsq) {
  __shared__ __align__(16) bf16_t sW[HD * WS];         // W1^T row-major
  __shared__ __align__(16) bf16_t sT[8][16 * TS];      // per-wave act tiles
  __shared__ float sB[HD];
  __shared__ float sSum[HD], sSq[HD];

  int tid = threadIdx.x;
  for (int t = tid; t < HD * HD; t += 256) {           // stage W1^T as bf16
    int k = t >> 7, n = t & 127;
    sW[n * WS + k] = (bf16_t)W1[t];
  }
  if (tid < HD) { sB[tid] = b1[tid]; sSum[tid] = 0.f; sSq[tid] = 0.f; }
  __syncthreads();

  int wave = tid >> 5, lane = tid & 31;
  size_t rowbase = (size_t)blockIdx.x * 128 + wave * 16;
  bf16_t* tile = sT[wave];

  // load 16x128 f32 rows -> bf16 LDS tile (zeros past N)
  for (int t = lane; t < 16 * 32; t += 32) {
    int m = t >> 5, c4 = t & 31;
    size_t r = rowbase + m;
    float4 v = make_float4(0.f, 0.f, 0.f, 0.f);
    if (r < NNODES) v = ((const float4*)agg)[r * 32 + c4];
    uint2 p; p.x = pack2(v.x, v.y); p.y = pack2(v.z, v.w);
    *(uint2*)&tile[m * TS + c4 * 4] = p;
  }
  __syncthreads();

  int nloc = lane & 15;
  int ka0  = (lane & 16) ? 8 : 0;
  int kb0  = (lane & 16) ? 8 : 0;
  int cb0  = (lane & 16) ? 8 : 0;
  size_t row  = rowbase + nloc;     // C layout: lane's column = node row
  bool valid  = row < NNODES;

#pragma unroll
  for (int nt = 0; nt < 8; ++nt) {
    v8f c = gemm_tile(&sW[(nt * 16 + nloc) * WS], &tile[nloc * TS], ka0, kb0);
    int ch = nt * 16 + cb0;
    float f[8];
#pragma unroll
    for (int r8 = 0; r8 < 8; ++r8) f[r8] = c[r8] + sB[ch + r8];
    if (valid) {
      uint4 p;
      p.x = pack2(f[0], f[1]); p.y = pack2(f[2], f[3]);
      p.z = pack2(f[4], f[5]); p.w = pack2(f[6], f[7]);
      *(uint4*)&h1[row * HD + ch] = p;
    }
    // BN stats: reduce over the 16 node-rows held by each half-wave
    float sv[8], sq[8];
#pragma unroll
    for (int r8 = 0; r8 < 8; ++r8) {
      float v = valid ? f[r8] : 0.f;
      sv[r8] = v; sq[r8] = v * v;
    }
#pragma unroll
    for (int off = 8; off >= 1; off >>= 1) {
#pragma unroll
      for (int r8 = 0; r8 < 8; ++r8) {
        sv[r8] += __shfl_xor(sv[r8], off, 32);
        sq[r8] += __shfl_xor(sq[r8], off, 32);
      }
    }
    if (nloc == 0) {
#pragma unroll
      for (int r8 = 0; r8 < 8; ++r8) {
        atomicAdd(&sSum[ch + r8], sv[r8]);
        atomicAdd(&sSq[ch + r8], sq[r8]);
      }
    }
  }
  __syncthreads();
  if (tid < HD) { atomicAdd(&gsum[tid], sSum[tid]); atomicAdd(&gsq[tid], sSq[tid]); }
}

// ---------------------------------------------------------------------------
// Kernel 3: BN params  scale = gamma*rsqrt(var+eps), shift = beta - mean*scale
// ---------------------------------------------------------------------------
__global__ void k_bn(const float* __restrict__ stats,   // [sum|sumsq|scale|shift]
                     const float* __restrict__ gamma,
                     const float* __restrict__ beta,
                     float* __restrict__ scale,
                     float* __restrict__ shift) {
  int t = threadIdx.x;
  float mean = stats[t] * (1.0f / NNODES);
  float var  = stats[HD + t] * (1.0f / NNODES) - mean * mean;
  float sc   = gamma[t] * rsqrtf(var + BN_EPS);
  scale[t] = sc;
  shift[t] = beta[t] - mean * sc;
}

// ---------------------------------------------------------------------------
// Kernel 4: fused  out = ( relu( relu( relu(BN(h1)) @W2+b2 ) @Wm1+bm1 ) @Wm2+bm2 )
// ---------------------------------------------------------------------------
__global__ __launch_bounds__(256) void k_mlp(const bf16_t* __restrict__ h1,
                                             const float* __restrict__ W2,
                                             const float* __restrict__ b2,
                                             const float* __restrict__ Wm1,
                                             const float* __restrict__ bm1,
                                             const float* __restrict__ Wm2,
                                             const float* __restrict__ bm2,
                                             const float* __restrict__ scale,
                                             const float* __restrict__ shift,
                                             float* __restrict__ out) {
  __shared__ __align__(16) bf16_t sW2[HD * WS];          // W2^T
  __shared__ __align__(16) bf16_t sWm1[HD * WS];         // Wm1^T
  __shared__ __align__(16) bf16_t sWm2[OD * WS];         // Wm2^T
  __shared__ __align__(16) bf16_t sTile[8 * 2 * 16 * TS];// ping-pong per wave
  __shared__ float sB2[HD], sBm1[HD], sBm2[OD], sScale[HD], sShift[HD];

  int tid = threadIdx.x;
  for (int t = tid; t < HD * HD; t += 256) {
    int k = t >> 7, n = t & 127;
    sW2[n * WS + k]  = (bf16_t)W2[t];
    sWm1[n * WS + k] = (bf16_t)Wm1[t];
  }
  for (int t = tid; t < HD * OD; t += 256) {
    int k = t >> 6, n = t & 63;
    sWm2[n * WS + k] = (bf16_t)Wm2[t];
  }
  if (tid < HD) { sB2[tid] = b2[tid]; sBm1[tid] = bm1[tid];
                  sScale[tid] = scale[tid]; sShift[tid] = shift[tid]; }
  if (tid < OD)   sBm2[tid] = bm2[tid];
  __syncthreads();

  int wave = tid >> 5, lane = tid & 31;
  size_t rowbase = (size_t)blockIdx.x * 128 + wave * 16;
  bf16_t* tileA = &sTile[wave * 2 * 16 * TS];
  bf16_t* tileB = tileA + 16 * TS;

  // stage h1 tile with BN affine + ReLU (bf16 in, bf16 out)
  for (int t = lane; t < 16 * 16; t += 32) {     // 256 chunks of 8 bf16
    int m = t >> 4, c8 = t & 15;
    size_t r = rowbase + m;
    uint4 p = make_uint4(0u, 0u, 0u, 0u);
    if (r < NNODES) {
      uint4 raw = *(const uint4*)&h1[r * HD + c8 * 8];
      float f[8]; unpack8(raw, f);
      int ch = c8 * 8;
#pragma unroll
      for (int j = 0; j < 8; ++j)
        f[j] = fmaxf(f[j] * sScale[ch + j] + sShift[ch + j], 0.f);
      p.x = pack2(f[0], f[1]); p.y = pack2(f[2], f[3]);
      p.z = pack2(f[4], f[5]); p.w = pack2(f[6], f[7]);
    }
    *(uint4*)&tileA[m * TS + c8 * 8] = p;
  }
  // Per-wave tiles are private; LDS ops are in-order within a wave.

  int nloc = lane & 15;
  int ka0  = (lane & 16) ? 8 : 0;
  int kb0  = (lane & 16) ? 8 : 0;
  int cb0  = (lane & 16) ? 8 : 0;

  // layer: tileA @ W2 + b2 -> relu -> tileB
#pragma unroll
  for (int nt = 0; nt < 8; ++nt) {
    v8f c = gemm_tile(&sW2[(nt * 16 + nloc) * WS], &tileA[nloc * TS], ka0, kb0);
    int ch = nt * 16 + cb0;
    float f[8];
#pragma unroll
    for (int r8 = 0; r8 < 8; ++r8) f[r8] = fmaxf(c[r8] + sB2[ch + r8], 0.f);
    uint4 p;
    p.x = pack2(f[0], f[1]); p.y = pack2(f[2], f[3]);
    p.z = pack2(f[4], f[5]); p.w = pack2(f[6], f[7]);
    *(uint4*)&tileB[nloc * TS + ch] = p;
  }

  // layer: tileB @ Wm1 + bm1 -> relu -> tileA
#pragma unroll
  for (int nt = 0; nt < 8; ++nt) {
    v8f c = gemm_tile(&sWm1[(nt * 16 + nloc) * WS], &tileB[nloc * TS], ka0, kb0);
    int ch = nt * 16 + cb0;
    float f[8];
#pragma unroll
    for (int r8 = 0; r8 < 8; ++r8) f[r8] = fmaxf(c[r8] + sBm1[ch + r8], 0.f);
    uint4 p;
    p.x = pack2(f[0], f[1]); p.y = pack2(f[2], f[3]);
    p.z = pack2(f[4], f[5]); p.w = pack2(f[6], f[7]);
    *(uint4*)&tileA[nloc * TS + ch] = p;
  }

  // layer: tileA @ Wm2 + bm2 -> out (f32, OUT=64 -> 4 channel tiles)
  size_t row  = rowbase + nloc;
  bool valid  = row < NNODES;
#pragma unroll
  for (int nt = 0; nt < 4; ++nt) {
    v8f c = gemm_tile(&sWm2[(nt * 16 + nloc) * WS], &tileA[nloc * TS], ka0, kb0);
    int ch = nt * 16 + cb0;
    if (valid) {
      float4 f0 = make_float4(c[0] + sBm2[ch + 0], c[1] + sBm2[ch + 1],
                              c[2] + sBm2[ch + 2], c[3] + sBm2[ch + 3]);
      float4 f1 = make_float4(c[4] + sBm2[ch + 4], c[5] + sBm2[ch + 5],
                              c[6] + sBm2[ch + 6], c[7] + sBm2[ch + 7]);
      *(float4*)&out[row * OD + ch]     = f0;
      *(float4*)&out[row * OD + ch + 4] = f1;
    }
  }
}

// ---------------------------------------------------------------------------
extern "C" void kernel_launch(void* const* d_in, const int* in_sizes, int n_in,
                              void* d_out, int out_size, void* d_ws, size_t ws_size,
                              hipStream_t stream) {
  const float*     x     = (const float*)d_in[0];
  const long long* ei    = (const long long*)d_in[1];   // int64 [2, E]
  const float*     W1    = (const float*)d_in[2];
  const float*     b1    = (const float*)d_in[3];
  const float*     gamma = (const float*)d_in[4];
  const float*     beta  = (const float*)d_in[5];
  const float*     W2    = (const float*)d_in[6];
  const float*     b2    = (const float*)d_in[7];
  const float*     Wm1   = (const float*)d_in[8];
  const float*     bm1   = (const float*)d_in[9];
  const float*     Wm2   = (const float*)d_in[10];
  const float*     bm2   = (const float*)d_in[11];
  float*           out   = (float*)d_out;

  // workspace layout
  const size_t AGG_BYTES = (size_t)NNODES * HD * sizeof(float);   // 51.2 MB
  const size_t H1_BYTES  = (size_t)NNODES * HD * sizeof(bf16_t);  // 25.6 MB
  float*  agg   = (float*)d_ws;
  bf16_t* h1    = (bf16_t*)((char*)d_ws + AGG_BYTES);
  float*  stats = (float*)((char*)d_ws + AGG_BYTES + H1_BYTES);
  // stats: [0,128) colsum | [128,256) colsumsq | [256,384) scale | [384,512) shift

  k_init   <<<(NNODES * 32) / 256, 256, 0, stream>>>(x, agg);
  k_zero   <<<1, 256, 0, stream>>>(stats);
  k_scatter<<<(NEDGES * 32) / 256, 256, 0, stream>>>(ei, x, agg);
  k_gemm1  <<<(NNODES + 127) / 128, 256, 0, stream>>>(agg, W1, b1, h1,
                                                      stats, stats + HD);
  k_bn     <<<1, HD, 0, stream>>>(stats, gamma, beta, stats + 256, stats + 384);
  k_mlp    <<<(NNODES + 127) / 128, 256, 0, stream>>>(h1, W2, b2, Wm1, bm1,
                                                      Wm2, bm2, stats + 256,
                                                      stats + 384, out);
}
